// TopkCFMultiHeadAttention_66803921322197
// MI455X (gfx1250) — matrix operand order repaired and checked by reading, not compiled, not run on hardware
//
#include <hip/hip_runtime.h>
#include <cstdint>
#include <cstddef>

typedef __attribute__((ext_vector_type(16))) _Float16 v16h;
typedef __attribute__((ext_vector_type(8)))  _Float16 h8;
typedef __attribute__((ext_vector_type(4)))  _Float16 h4;
typedef __attribute__((ext_vector_type(8)))  float    v8f;
typedef int i4 __attribute__((vector_size(16)));     // matches builtin prototype

#define S_   16384
#define IN_  1024
#define E_   512
#define H_   8
#define D_   64
#define C_   8
#define K_   16

// ---------------------------------------------------------------------------
// CDNA5 async global->LDS copy (ASYNCcnt path), with synchronous fallback.
// ROCm clang prototype: (AS1 i4* gsrc, AS3 i4* ldst, imm offset, imm cpol)
// ---------------------------------------------------------------------------
#if defined(__has_builtin)
#if __has_builtin(__builtin_amdgcn_global_load_async_to_lds_b128)
#define HAVE_ASYNC_LDS 1
#endif
#endif

typedef __attribute__((address_space(1))) i4 gi4;
typedef __attribute__((address_space(3))) i4 li4;

__device__ __forceinline__ void cp16_async(void* lds, const void* gp) {
#ifdef HAVE_ASYNC_LDS
    __builtin_amdgcn_global_load_async_to_lds_b128((gi4*)gp, (li4*)lds, 0, 0);
#else
    *(h8*)lds = *(const h8*)gp;
#endif
}

__device__ __forceinline__ void wait_async_all() {
#ifdef HAVE_ASYNC_LDS
#if __has_builtin(__builtin_amdgcn_s_wait_asynccnt)
    __builtin_amdgcn_s_wait_asynccnt(0);
#else
    asm volatile("s_wait_asynccnt 0x0" ::: "memory");
#endif
#endif
}

// ---------------------------------------------------------------------------
// f32 -> f16 conversion (grid-stride)
// ---------------------------------------------------------------------------
__global__ void cvt_f16_kernel(const float* __restrict__ a, _Float16* __restrict__ o, int n) {
    for (int i = blockIdx.x * 256 + threadIdx.x; i < n; i += gridDim.x * 256)
        o[i] = (_Float16)a[i];
}

// ---------------------------------------------------------------------------
// Fused: c[s,:] = x[s,:] @ W_ih.T + b_ih   and   xh[s,:] = f16(x[s,:])
// ---------------------------------------------------------------------------
__global__ __launch_bounds__(256)
void score_convx_kernel(const float* __restrict__ x, const float* __restrict__ Wih,
                        const float* __restrict__ bih, float* __restrict__ cout,
                        _Float16* __restrict__ xh) {
    const int s = blockIdx.x;
    const int t = threadIdx.x;
    const float* xr = x + (size_t)s * IN_;
    float4 xv = ((const float4*)xr)[t];           // covers 1024 floats with 256 thr

    h4 hx; hx[0] = (_Float16)xv.x; hx[1] = (_Float16)xv.y;
           hx[2] = (_Float16)xv.z; hx[3] = (_Float16)xv.w;
    ((h4*)(xh + (size_t)s * IN_))[t] = hx;

    float part[C_];
#pragma unroll
    for (int c = 0; c < C_; ++c) {
        float4 wv = ((const float4*)(Wih + (size_t)c * IN_))[t];
        part[c] = xv.x * wv.x + xv.y * wv.y + xv.z * wv.z + xv.w * wv.w;
    }
    __shared__ float red[256];
#pragma unroll
    for (int c = 0; c < C_; ++c) {
        red[t] = part[c];
        __syncthreads();
        for (int off = 128; off > 0; off >>= 1) {
            if (t < off) red[t] += red[t + off];
            __syncthreads();
        }
        if (t == 0) cout[(size_t)s * C_ + c] = red[0] + bih[c];
        __syncthreads();
    }
}

// ---------------------------------------------------------------------------
// Iterative top-16 per class column (value desc, then lower index).
// ---------------------------------------------------------------------------
__global__ __launch_bounds__(256)
void topk_kernel(const float* __restrict__ cbuf, int* __restrict__ idx_i,
                 float* __restrict__ idx_f) {
    const int cls = blockIdx.x;
    const int t = threadIdx.x;
    __shared__ float bval[256];
    __shared__ int   bidx[256];
    __shared__ int   chosen[K_];

    for (int r = 0; r < K_; ++r) {
        float best = -__builtin_inff();
        int besti = 0x7fffffff;
        for (int s = t; s < S_; s += 256) {
            bool skip = false;
            for (int q = 0; q < r; ++q)
                if (chosen[q] == s) { skip = true; break; }
            if (skip) continue;
            float v = cbuf[(size_t)s * C_ + cls];
            if (v > best || (v == best && s < besti)) { best = v; besti = s; }
        }
        bval[t] = best; bidx[t] = besti;
        __syncthreads();
        for (int off = 128; off > 0; off >>= 1) {
            if (t < off) {
                float v2 = bval[t + off]; int i2 = bidx[t + off];
                if (v2 > bval[t] || (v2 == bval[t] && i2 < bidx[t])) { bval[t] = v2; bidx[t] = i2; }
            }
            __syncthreads();
        }
        if (t == 0) {
            chosen[r] = bidx[0];
            idx_i[r * C_ + cls] = bidx[0];
            idx_f[r * C_ + cls] = (float)bidx[0];
        }
        __syncthreads();
    }
}

// ---------------------------------------------------------------------------
// k-projection: kraw[c,e] = sum_{kk,ii} x[idx[kk,c],ii] * W_k[e, kk*IN+ii] + b_k[e]
// ---------------------------------------------------------------------------
__global__ __launch_bounds__(256)
void kproj_kernel(const float* __restrict__ x, const float* __restrict__ Wk,
                  const float* __restrict__ bk, const int* __restrict__ idx,
                  float* __restrict__ kraw) {
    const int wg   = blockIdx.x * 8 + (threadIdx.x >> 5);   // 0..4095
    const int lane = threadIdx.x & 31;
    const int c = wg >> 9;
    const int e = wg & (E_ - 1);
    float acc = 0.f;
    const float* wrow = Wk + (size_t)e * (K_ * IN_);
#pragma unroll 1
    for (int kk = 0; kk < K_; ++kk) {
        const size_t xb = (size_t)idx[kk * C_ + c] * IN_;
        const float* wseg = wrow + kk * IN_;
        for (int ii = lane; ii < IN_; ii += 32)
            acc += x[xb + ii] * wseg[ii];
    }
#pragma unroll
    for (int m = 1; m < 32; m <<= 1) acc += __shfl_xor(acc, m, 32);
    if (lane == 0) kraw[(size_t)c * E_ + e] = acc + bk[e];
}

// ---------------------------------------------------------------------------
// Row LayerNorm (one block per row; supports in-place).
// ---------------------------------------------------------------------------
__global__ __launch_bounds__(256)
void ln_kernel(const float* __restrict__ in, float* __restrict__ out,
               const float* __restrict__ g, const float* __restrict__ b, int width) {
    const int row = blockIdx.x;
    const int t = threadIdx.x;
    const float* x = in + (size_t)row * width;
    float s = 0.f, s2 = 0.f;
    for (int i = t; i < width; i += 256) { float v = x[i]; s += v; s2 += v * v; }
    __shared__ float r1[256], r2[256];
    r1[t] = s; r2[t] = s2;
    __syncthreads();
    for (int off = 128; off > 0; off >>= 1) {
        if (t < off) { r1[t] += r1[t + off]; r2[t] += r2[t + off]; }
        __syncthreads();
    }
    const float mu  = r1[0] / width;
    const float var = r2[0] / width - mu * mu;
    const float inv = rsqrtf(var + 1e-5f);
    float* y = out + (size_t)row * width;
    for (int i = t; i < width; i += 256)
        y[i] = (x[i] - mu) * inv * g[i] + b[i];
}

// ---------------------------------------------------------------------------
// WMMA projection GEMM: Y[S,512] = Xh[S,1024] @ Wh[512,1024]^T + bias
// Block tile 128(M) x 64(N), K-step 32, f16 in / f32 accum.
// 8 waves: 4(M) x 2(N), each wave -> 2x2 of 16x16 v_wmma_f32_16x16x32_f16.
// Double-buffered LDS staged via global_load_async_to_lds_b128 (ASYNCcnt),
// so tile kt+1 streams from HBM while tile kt feeds the matrix pipes.
// LDS row stride 40 halves (80 B) => 16-B aligned b128, conflict-free.
// ---------------------------------------------------------------------------
#define PAD_ 40
#define NKT_ (IN_ / 32)
__global__ __launch_bounds__(256)
void proj_gemm_kernel(const _Float16* __restrict__ X, const _Float16* __restrict__ W,
                      const float* __restrict__ bias, float* __restrict__ Y) {
    const int tid  = threadIdx.x;
    const int lane = tid & 31, wave = tid >> 5;
    const int wm = wave & 3, wn = wave >> 2;        // 4x2 wave grid
    const int r  = lane & 15, g = lane >> 4;
    const int m0 = blockIdx.y * 128;
    const int n0 = blockIdx.x * 64;

    __shared__ _Float16 As[2][128 * PAD_];
    __shared__ _Float16 Bs[2][64 * PAD_];

    const v8f vzero = {0.f,0.f,0.f,0.f,0.f,0.f,0.f,0.f};
    v8f acc[2][2];
#pragma unroll
    for (int mi = 0; mi < 2; ++mi)
#pragma unroll
        for (int ni = 0; ni < 2; ++ni) acc[mi][ni] = vzero;

    const int rowa = tid >> 2;            // 0..63
    const int col8 = (tid & 3) * 8;       // 0,8,16,24

    // stage tile kt into LDS buffer b (3 x b128 per thread, async)
    auto stage = [&](int b, int k0) {
        cp16_async(&As[b][rowa * PAD_ + col8],
                   &X[(size_t)(m0 + rowa) * IN_ + k0 + col8]);
        cp16_async(&As[b][(rowa + 64) * PAD_ + col8],
                   &X[(size_t)(m0 + rowa + 64) * IN_ + k0 + col8]);
        cp16_async(&Bs[b][rowa * PAD_ + col8],
                   &W[(size_t)(n0 + rowa) * IN_ + k0 + col8]);
    };

    stage(0, 0);
    wait_async_all();
    __syncthreads();

#pragma unroll 1
    for (int kt = 0; kt < NKT_; ++kt) {
        const int cur = kt & 1;
        if (kt + 1 < NKT_) {
            stage(cur ^ 1, (kt + 1) * 32);               // overlap with compute
            __builtin_prefetch(&X[(size_t)(m0 + rowa) * IN_ + (kt + 2) * 32 + col8], 0, 3);
        }

        // A fragment: lane(r,g) holds k = {8g..8g+7, 8g+16..8g+23}  (ISA 16-bit A 16x32)
        v16h afrag[2], bfrag[2];
#pragma unroll
        for (int mi = 0; mi < 2; ++mi) {
            const int arow = wm * 32 + mi * 16 + r;
            const h8 lo = *(const h8*)&As[cur][arow * PAD_ + 8 * g];
            const h8 hi = *(const h8*)&As[cur][arow * PAD_ + 8 * g + 16];
#pragma unroll
            for (int i = 0; i < 8; ++i) { afrag[mi][i] = lo[i]; afrag[mi][i + 8] = hi[i]; }
        }
        // B fragment: lane(n=r,g) holds k = {8g..8g+7, 8g+16..8g+23}  (ISA 16-bit B layout)
#pragma unroll
        for (int ni = 0; ni < 2; ++ni) {
            const int brow = wn * 32 + ni * 16 + r;
            const h8 lo = *(const h8*)&Bs[cur][brow * PAD_ + 8 * g];
            const h8 hi = *(const h8*)&Bs[cur][brow * PAD_ + 8 * g + 16];
#pragma unroll
            for (int i = 0; i < 8; ++i) { bfrag[ni][i] = lo[i]; bfrag[ni][i + 8] = hi[i]; }
        }
#pragma unroll
        for (int mi = 0; mi < 2; ++mi)
#pragma unroll
            for (int ni = 0; ni < 2; ++ni)
                acc[mi][ni] = __builtin_amdgcn_wmma_f32_16x16x32_f16(
                    false, afrag[mi], false, bfrag[ni],
                    (short)0, acc[mi][ni], false, false);

        if (kt + 1 < NKT_) wait_async_all();             // next tile landed
        __syncthreads();                                 // all waves done with cur
    }

    // D layout: element j -> row j + 8*(lane>>4), col lane&15
    const int ccol = lane & 15, hi2 = lane >> 4;
#pragma unroll
    for (int mi = 0; mi < 2; ++mi)
#pragma unroll
        for (int ni = 0; ni < 2; ++ni) {
            const int gn = n0 + wn * 32 + ni * 16 + ccol;
            const float bv = bias[gn];
            const int gm = m0 + wm * 32 + mi * 16 + 8 * hi2;
#pragma unroll
            for (int j = 0; j < 8; ++j)
                Y[(size_t)(gm + j) * E_ + gn] = acc[mi][ni][j] + bv;
        }
}

// ---------------------------------------------------------------------------
// Fused attention: logits=q·k/8, softmax over C=8, write attn[H,S,C],
// accumulate vals[h,c,d] += attn*v via atomics.
// ---------------------------------------------------------------------------
__global__ __launch_bounds__(256)
void attn_kernel(const float* __restrict__ q, const float* __restrict__ v,
                 const float* __restrict__ kln, float* __restrict__ attn_out,
                 float* __restrict__ vals) {
    const int h = blockIdx.x;
    const int lane = threadIdx.x & 31, wave = threadIdx.x >> 5;
    __shared__ float ks[C_ * D_];
    for (int i = threadIdx.x; i < C_ * D_; i += 256)
        ks[i] = kln[(size_t)(i >> 6) * E_ + h * D_ + (i & 63)];
    __syncthreads();

    float vp[C_][2];
#pragma unroll
    for (int c = 0; c < C_; ++c) { vp[c][0] = 0.f; vp[c][1] = 0.f; }

    const int s0 = blockIdx.y * 1024;
    for (int s = s0 + wave; s < s0 + 1024; s += 8) {
        const size_t base = (size_t)s * E_ + h * D_;
        const float q0 = q[base + lane], q1 = q[base + lane + 32];
        const float v0 = v[base + lane], v1 = v[base + lane + 32];
        float ex[C_];
        float mx = -__builtin_inff();
#pragma unroll
        for (int c = 0; c < C_; ++c) {
            float p = q0 * ks[c * D_ + lane] + q1 * ks[c * D_ + lane + 32];
#pragma unroll
            for (int m = 1; m < 32; m <<= 1) p += __shfl_xor(p, m, 32);
            ex[c] = p * 0.125f;
            mx = fmaxf(mx, ex[c]);
        }
        float se = 0.f;
#pragma unroll
        for (int c = 0; c < C_; ++c) { ex[c] = __expf(ex[c] - mx); se += ex[c]; }
        const float inv = 1.f / se;
        float mine = 0.f;
#pragma unroll
        for (int c = 0; c < C_; ++c) {
            ex[c] *= inv;
            vp[c][0] += ex[c] * v0;
            vp[c][1] += ex[c] * v1;
            mine = (lane == c) ? ex[c] : mine;
        }
        if (lane < C_) attn_out[((size_t)h * S_ + s) * C_ + lane] = mine;
    }
#pragma unroll
    for (int c = 0; c < C_; ++c) {
        atomicAdd(&vals[(size_t)(h * C_ + c) * D_ + lane],      vp[c][0]);
        atomicAdd(&vals[(size_t)(h * C_ + c) * D_ + lane + 32], vp[c][1]);
    }
}

// ---------------------------------------------------------------------------
// Fused out-projection + LN: o[c,:] = LN(vals_row(c) @ W_p.T + b_p)
// ---------------------------------------------------------------------------
__global__ __launch_bounds__(256)
void outproj_kernel(const float* __restrict__ vals, const float* __restrict__ Wp,
                    const float* __restrict__ bp, const float* __restrict__ g,
                    const float* __restrict__ be, float* __restrict__ o) {
    const int c = blockIdx.x;
    const int t = threadIdx.x;
    __shared__ float vrow[E_];
    __shared__ float orow[IN_];
    __shared__ float r1[256], r2[256];

    for (int e = t; e < E_; e += 256)                // [C,E] row = vals[h=e>>6][c][e&63]
        vrow[e] = vals[(size_t)((e >> 6) * C_ + c) * D_ + (e & 63)];
    __syncthreads();

    for (int j = t; j < IN_; j += 256) {
        float acc = bp[j];
        const float* w = Wp + (size_t)j * E_;
        for (int e = 0; e < E_; ++e) acc += vrow[e] * w[e];
        orow[j] = acc;
    }
    __syncthreads();

    float s = 0.f, s2 = 0.f;
    for (int j = t; j < IN_; j += 256) { float x = orow[j]; s += x; s2 += x * x; }
    r1[t] = s; r2[t] = s2;
    __syncthreads();
    for (int off = 128; off > 0; off >>= 1) {
        if (t < off) { r1[t] += r1[t + off]; r2[t] += r2[t + off]; }
        __syncthreads();
    }
    const float mu  = r1[0] / IN_;
    const float var = r2[0] / IN_ - mu * mu;
    const float inv = rsqrtf(var + 1e-5f);
    for (int j = t; j < IN_; j += 256)
        o[(size_t)c * IN_ + j] = (orow[j] - mu) * inv * g[j] + be[j];
}

// ---------------------------------------------------------------------------
extern "C" void kernel_launch(void* const* d_in, const int* in_sizes, int n_in,
                              void* d_out, int out_size, void* d_ws, size_t ws_size,
                              hipStream_t stream) {
    (void)in_sizes; (void)n_in; (void)out_size; (void)ws_size;

    const float* x    = (const float*)d_in[0];
    const float* W_ih = (const float*)d_in[1];
    const float* b_ih = (const float*)d_in[2];
    const float* W_q  = (const float*)d_in[3];
    const float* b_q  = (const float*)d_in[4];
    const float* g_q  = (const float*)d_in[5];
    const float* be_q = (const float*)d_in[6];
    const float* W_k  = (const float*)d_in[7];
    const float* b_k  = (const float*)d_in[8];
    const float* g_k  = (const float*)d_in[9];
    const float* be_k = (const float*)d_in[10];
    const float* W_v  = (const float*)d_in[11];
    const float* b_v  = (const float*)d_in[12];
    const float* g_v  = (const float*)d_in[13];
    const float* be_v = (const float*)d_in[14];
    const float* W_p  = (const float*)d_in[15];
    const float* b_p  = (const float*)d_in[16];
    const float* g_o  = (const float*)d_in[17];
    const float* be_o = (const float*)d_in[18];

    // outputs: o[C,IN] | c[S,C] | attn[H,S,C] | topk_idx[K,C] (as float)
    float* out_o    = (float*)d_out;
    float* out_c    = out_o + C_ * IN_;                 // +8192
    float* out_attn = out_c + (size_t)S_ * C_;          // +139264
    float* out_idx  = out_attn + (size_t)H_ * S_ * C_;  // +1187840

    // workspace layout (bytes)
    char* ws = (char*)d_ws;
    _Float16* xh    = (_Float16*)(ws);                               // 32 MB
    _Float16* wqh   = (_Float16*)(ws + 33554432);                    // 1 MB
    _Float16* wvh   = (_Float16*)(ws + 34603008);                    // 1 MB
    float*    q_raw = (float*)   (ws + 35651584);                    // 32 MB
    float*    v_raw = (float*)   (ws + 69206016);                    // 32 MB
    float*    kraw  = (float*)   (ws + 102760448);                   // 16 KB
    float*    kln   = (float*)   (ws + 102776832);                   // 16 KB
    float*    vals  = (float*)   (ws + 102793216);                   // 16 KB
    int*      idx_i = (int*)     (ws + 102809600);                   // 512 B

    (void)hipMemsetAsync(vals, 0, (size_t)H_ * C_ * D_ * sizeof(float), stream);

    // weight conversion to f16
    cvt_f16_kernel<<<512, 256, 0, stream>>>(W_q, wqh, E_ * IN_);
    cvt_f16_kernel<<<512, 256, 0, stream>>>(W_v, wvh, E_ * IN_);

    // instance scores + x->f16
    score_convx_kernel<<<S_, 256, 0, stream>>>(x, W_ih, b_ih, out_c, xh);

    // top-16 per class
    topk_kernel<<<C_, 256, 0, stream>>>(out_c, idx_i, out_idx);

    // k projection + LN
    kproj_kernel<<<(C_ * E_) / 8, 256, 0, stream>>>(x, W_k, b_k, idx_i, kraw);
    ln_kernel<<<C_, 256, 0, stream>>>(kraw, kln, g_k, be_k, E_);

    // q / v projections (WMMA + async LDS pipeline) + in-place row LN
    proj_gemm_kernel<<<dim3(E_ / 64, S_ / 128), 256, 0, stream>>>(xh, wqh, b_q, q_raw);
    proj_gemm_kernel<<<dim3(E_ / 64, S_ / 128), 256, 0, stream>>>(xh, wvh, b_v, v_raw);
    ln_kernel<<<S_, 256, 0, stream>>>(q_raw, q_raw, g_q, be_q, E_);
    ln_kernel<<<S_, 256, 0, stream>>>(v_raw, v_raw, g_v, be_v, E_);

    // attention (logits + softmax + attn write + attn^T v)
    attn_kernel<<<dim3(H_, S_ / 1024), 256, 0, stream>>>(q_raw, v_raw, kln, out_attn, vals);

    // output projection + LN
    outproj_kernel<<<C_, 256, 0, stream>>>(vals, W_p, b_p, g_o, be_o, out_o);
}
